// MoETSAEncoderLayer_46385646797316
// MI455X (gfx1250) — compile-verified, matrix operand order checked by reading, not run on hardware
//
#include <hip/hip_runtime.h>
#include <hip/hip_bf16.h>
#include <cstdint>

#define B_   8
#define L_   1024
#define D_   1024
#define H_   16
#define DFF_ 4096
#define E_   8
#define DC_  512
#define DH_  32
#define N_   (B_*L_)

typedef __attribute__((ext_vector_type(16))) __bf16 v16bf;
typedef __attribute__((ext_vector_type(8)))  float  v8f;

static __device__ __forceinline__ v8f wmma_bf16(v16bf a, v16bf b, v8f c) {
  return __builtin_amdgcn_wmma_f32_16x16x32_bf16(false, a, false, b, (short)0, c, false, false);
}

// Low 32 bits of the flat shared-aperture address = LDS byte offset.
static __device__ __forceinline__ uint32_t lds_addr(const void* p) {
  return (uint32_t)(uintptr_t)p;
}

// CDNA5 async copy: global -> LDS, 16 bytes per lane, tracked by ASYNCcnt.
static __device__ __forceinline__ void async_copy_b128(uint32_t dst_lds, const void* src) {
  asm volatile("global_load_async_to_lds_b128 %0, %1, off"
               :: "v"(dst_lds), "v"(src) : "memory");
}
static __device__ __forceinline__ void wait_async0() {
  asm volatile("s_wait_asynccnt 0x0" ::: "memory");
}

// ---------------------------------------------------------------------------
// f32 -> bf16 conversion (grid-stride)
// ---------------------------------------------------------------------------
__global__ __launch_bounds__(256) void cvt_f32_bf16(const float* __restrict__ x,
                                                    __bf16* __restrict__ y, long n) {
  long i = (long)blockIdx.x * blockDim.x + threadIdx.x;
  long stride = (long)gridDim.x * blockDim.x;
  for (; i < n; i += stride) y[i] = (__bf16)x[i];
}

// pack x1 (left half of src row) into left half of cat buffer (ld = D)
__global__ __launch_bounds__(256) void pack_x1_kernel(const float* __restrict__ src,
                                                      __bf16* __restrict__ catb) {
  size_t i = (size_t)blockIdx.x * blockDim.x + threadIdx.x;
  if (i >= (size_t)N_ * DC_) return;
  size_t n = i >> 9; int c = (int)(i & (DC_ - 1));
  catb[n * D_ + c] = (__bf16)src[n * D_ + c];
}

// ---------------------------------------------------------------------------
// Generic bf16 WMMA GEMM: C[M,Nc] = A[M,Kd] * B[Kd,Nc]  (row-major, strides)
// EPI 0: outF = acc + bias      (f32)
// EPI 1: outB = bf16(acc+bias)
// EPI 2: outF += (acc+bias) * rowScale[row]   (MoE accumulate)
// Block tile 128x64, 8 waves -> each wave 32 rows x 32 cols (4 WMMAs), K step 32.
// Tiles staged with GLOBAL_LOAD_ASYNC_TO_LDS_B128 (ASYNCcnt).
// LDS pitches: A 40 bf16 (80B, 16B-aligned rows), B 72 bf16 (144B).
// ---------------------------------------------------------------------------
template <int EPI>
__global__ __launch_bounds__(256) void gemm_bf16_kernel(
    const __bf16* __restrict__ A, int lda,
    const __bf16* __restrict__ Bm, int ldb,
    const float* __restrict__ bias,
    float* __restrict__ outF, __bf16* __restrict__ outB, int ldc,
    const float* __restrict__ rowScale,
    int M, int Nc, int Kd) {
  (void)M; (void)Nc;
  __shared__ __bf16 As[128][40];
  __shared__ __bf16 Bs[32][72];
  const int m0 = blockIdx.x * 128;
  const int n0 = blockIdx.y * 64;
  const int tid = threadIdx.x;
  const int wave = tid >> 5, lane = tid & 31;
  const int wm = (wave >> 1) * 32;
  const int wn = (wave & 1) * 32;
  const int lr = lane & 15;
  const int hi8 = (lane >> 4) << 3;
  const uint32_t asb = lds_addr(&As[0][0]);
  const uint32_t bsb = lds_addr(&Bs[0][0]);

  v8f acc00 = {0.f,0.f,0.f,0.f,0.f,0.f,0.f,0.f};
  v8f acc01 = {0.f,0.f,0.f,0.f,0.f,0.f,0.f,0.f};
  v8f acc10 = {0.f,0.f,0.f,0.f,0.f,0.f,0.f,0.f};
  v8f acc11 = {0.f,0.f,0.f,0.f,0.f,0.f,0.f,0.f};

  for (int k0 = 0; k0 < Kd; k0 += 32) {
    if (k0 + 32 < Kd) {  // global_prefetch_b8 of the next K-slab
      __builtin_prefetch(&A[(size_t)(m0 + (tid >> 1)) * lda + k0 + 32], 0, 1);
      __builtin_prefetch(&Bm[(size_t)(k0 + 32 + (tid & 31)) * ldb + n0], 0, 1);
    }
    // A tile: 128 rows x 64B -> 512 x 16B chunks (2 per thread), async to LDS
    #pragma unroll
    for (int ch = 0; ch < 2; ++ch) {
      int i = tid + ch * 256;
      int r = i >> 2, cb = (i & 3) * 16;
      async_copy_b128(asb + (uint32_t)(r * 80 + cb),
                      (const char*)A + ((size_t)(m0 + r) * lda + k0) * 2 + cb);
    }
    // B tile: 32 rows x 128B -> 256 x 16B chunks (1 per thread)
    {
      int r = tid >> 3, cb = (tid & 7) * 16;
      async_copy_b128(bsb + (uint32_t)(r * 144 + cb),
                      (const char*)Bm + ((size_t)(k0 + r) * ldb + n0) * 2 + cb);
    }
    wait_async0();
    __syncthreads();

    v16bf a0, a1, b0, b1;
    #pragma unroll
    for (int v = 0; v < 16; ++v) {
      int kk = (v < 8 ? v : v + 8) + hi8;
      a0[v] = As[wm + lr][kk];
      a1[v] = As[wm + 16 + lr][kk];
      b0[v] = Bs[kk][wn + lr];
      b1[v] = Bs[kk][wn + 16 + lr];
    }
    acc00 = wmma_bf16(a0, b0, acc00);
    acc01 = wmma_bf16(a0, b1, acc01);
    acc10 = wmma_bf16(a1, b0, acc10);
    acc11 = wmma_bf16(a1, b1, acc11);
    __syncthreads();
  }

  #pragma unroll
  for (int r = 0; r < 8; ++r) {
    int row0 = m0 + wm + r + hi8;
    int row1 = row0 + 16;
    int c0 = n0 + wn + lr;
    int c1 = c0 + 16;
    float bv0 = bias ? bias[c0] : 0.f;
    float bv1 = bias ? bias[c1] : 0.f;
    float v00 = acc00[r] + bv0, v01 = acc01[r] + bv1;
    float v10 = acc10[r] + bv0, v11 = acc11[r] + bv1;
    if (EPI == 0) {
      outF[(size_t)row0 * ldc + c0] = v00;
      outF[(size_t)row0 * ldc + c1] = v01;
      outF[(size_t)row1 * ldc + c0] = v10;
      outF[(size_t)row1 * ldc + c1] = v11;
    } else if (EPI == 1) {
      outB[(size_t)row0 * ldc + c0] = (__bf16)v00;
      outB[(size_t)row0 * ldc + c1] = (__bf16)v01;
      outB[(size_t)row1 * ldc + c0] = (__bf16)v10;
      outB[(size_t)row1 * ldc + c1] = (__bf16)v11;
    } else {
      float g0 = rowScale[row0], g1 = rowScale[row1];
      outF[(size_t)row0 * ldc + c0] += v00 * g0;
      outF[(size_t)row0 * ldc + c1] += v01 * g0;
      outF[(size_t)row1 * ldc + c0] += v10 * g1;
      outF[(size_t)row1 * ldc + c1] += v11 * g1;
    }
  }
}

// ---------------------------------------------------------------------------
// Conv saliency: sal[b,h,l] = sum_{c,t} x1[b,l+t-1,c] * conv_w[h,c,t] + conv_b[h]
// ---------------------------------------------------------------------------
__global__ __launch_bounds__(256) void sal_kernel(const float* __restrict__ src,
                                                  const float* __restrict__ conv_w,
                                                  const float* __restrict__ conv_b,
                                                  float* __restrict__ sal) {
  int idx = blockIdx.x * blockDim.x + threadIdx.x;
  if (idx >= B_ * H_ * L_) return;
  int l = idx & (L_ - 1);
  int h = (idx >> 10) & (H_ - 1);
  int b = idx >> 14;
  float acc = conv_b[h];
  for (int t = 0; t < 3; ++t) {
    int ll = l + t - 1;
    if (ll < 0 || ll >= L_) continue;
    const float* xr = src + ((size_t)b * L_ + ll) * D_;
    const float* wr = conv_w + (size_t)h * DC_ * 3 + t;
    float s = 0.f;
    for (int c = 0; c < DC_; ++c) s += xr[c] * wr[(size_t)c * 3];
    acc += s;
  }
  sal[idx] = acc;
}

// ---------------------------------------------------------------------------
// Fused attention: per (b, h, 64-row q block).
// Pass 1: online softmax stats via WMMA score tiles. Pass 2: recompute scores,
// write normalized attn f32 to d_out, WMMA-accumulate ctx = P @ K (bf16 out).
// Q/K tiles staged with async global->LDS copies.
// ---------------------------------------------------------------------------
__global__ __launch_bounds__(256) void attn_kernel(const __bf16* __restrict__ Qb,
                                                   const __bf16* __restrict__ Kb,
                                                   const float* __restrict__ sal,
                                                   float* __restrict__ attn,
                                                   __bf16* __restrict__ ctxb) {
  const float scale = 0.17677669529663687f;  // 1/sqrt(32)
  const int q0 = blockIdx.x * 64;
  const int h  = blockIdx.y;
  const int b  = blockIdx.z;
  __shared__ __bf16 Qt[64][DH_];   // 64B rows, 16B aligned
  __shared__ __bf16 Kt[64][DH_];
  __shared__ float  Sc[64][68];
  __shared__ __bf16 Pt[64][64];
  __shared__ float  rm[64], rs[64];

  const int tid = threadIdx.x;
  const int wave = tid >> 5, lane = tid & 31;
  const int wm = (wave >> 1) * 16;
  const int wn = (wave & 1) * 32;
  const int lr = lane & 15;
  const int hi8 = (lane >> 4) << 3;
  const uint32_t qtb = lds_addr(&Qt[0][0]);
  const uint32_t ktb = lds_addr(&Kt[0][0]);

  // stage Q tile: 64 rows x 64B -> 256 x 16B chunks (1 per thread)
  {
    int r = tid >> 2, cb = (tid & 3) * 16;
    const char* gsrc = (const char*)(Qb + ((size_t)(b * L_ + q0)) * DC_ + h * DH_) +
                       (size_t)r * DC_ * 2 + cb;
    async_copy_b128(qtb + (uint32_t)(r * 64 + cb), gsrc);
  }
  if (tid < 64) { rm[tid] = -1e30f; rs[tid] = 0.f; }
  wait_async0();
  __syncthreads();

  const float* salrow = sal + ((size_t)b * H_ + h) * L_;

  // ---- pass 1: softmax stats ----
  for (int kt = 0; kt < L_ / 64; ++kt) {
    const int k0 = kt * 64;
    {
      int r = tid >> 2, cb = (tid & 3) * 16;
      const char* gsrc = (const char*)(Kb + ((size_t)(b * L_ + k0)) * DC_ + h * DH_) +
                         (size_t)r * DC_ * 2 + cb;
      async_copy_b128(ktb + (uint32_t)(r * 64 + cb), gsrc);
    }
    wait_async0();
    __syncthreads();
    v16bf a, b0, b1;
    #pragma unroll
    for (int v = 0; v < 16; ++v) {
      int kk = (v < 8 ? v : v + 8) + hi8;
      a[v]  = Qt[wm + lr][kk];
      b0[v] = Kt[wn + lr][kk];        // B[k=dh][n=key] = K[key][dh]
      b1[v] = Kt[wn + 16 + lr][kk];
    }
    v8f s0 = {0.f,0.f,0.f,0.f,0.f,0.f,0.f,0.f};
    v8f s1 = {0.f,0.f,0.f,0.f,0.f,0.f,0.f,0.f};
    s0 = wmma_bf16(a, b0, s0);
    s1 = wmma_bf16(a, b1, s1);
    #pragma unroll
    for (int r = 0; r < 8; ++r) {
      int row = wm + r + hi8;
      Sc[row][wn + lr]      = s0[r] * scale + salrow[k0 + wn + lr];
      Sc[row][wn + 16 + lr] = s1[r] * scale + salrow[k0 + wn + 16 + lr];
    }
    __syncthreads();
    if (tid < 64) {
      float mx = rm[tid];
      for (int c = 0; c < 64; ++c) mx = fmaxf(mx, Sc[tid][c]);
      float sum = rs[tid] * __expf(rm[tid] - mx);
      for (int c = 0; c < 64; ++c) sum += __expf(Sc[tid][c] - mx);
      rm[tid] = mx; rs[tid] = sum;
    }
    __syncthreads();
  }

  // ---- pass 2: write attn + accumulate ctx ----
  v8f ctx = {0.f,0.f,0.f,0.f,0.f,0.f,0.f,0.f};
  const int dh0 = (wave & 1) * 16;
  for (int kt = 0; kt < L_ / 64; ++kt) {
    const int k0 = kt * 64;
    {
      int r = tid >> 2, cb = (tid & 3) * 16;
      const char* gsrc = (const char*)(Kb + ((size_t)(b * L_ + k0)) * DC_ + h * DH_) +
                         (size_t)r * DC_ * 2 + cb;
      async_copy_b128(ktb + (uint32_t)(r * 64 + cb), gsrc);
    }
    wait_async0();
    __syncthreads();
    v16bf a, b0, b1;
    #pragma unroll
    for (int v = 0; v < 16; ++v) {
      int kk = (v < 8 ? v : v + 8) + hi8;
      a[v]  = Qt[wm + lr][kk];
      b0[v] = Kt[wn + lr][kk];
      b1[v] = Kt[wn + 16 + lr][kk];
    }
    v8f s0 = {0.f,0.f,0.f,0.f,0.f,0.f,0.f,0.f};
    v8f s1 = {0.f,0.f,0.f,0.f,0.f,0.f,0.f,0.f};
    s0 = wmma_bf16(a, b0, s0);
    s1 = wmma_bf16(a, b1, s1);
    #pragma unroll
    for (int r = 0; r < 8; ++r) {
      int row = wm + r + hi8;
      float inv = 1.f / rs[row];
      float p0 = __expf(s0[r] * scale + salrow[k0 + wn + lr]      - rm[row]) * inv;
      float p1 = __expf(s1[r] * scale + salrow[k0 + wn + 16 + lr] - rm[row]) * inv;
      size_t abase = (((size_t)b * H_ + h) * L_ + (size_t)(q0 + row)) * L_;
      attn[abase + k0 + wn + lr]      = p0;
      attn[abase + k0 + wn + 16 + lr] = p1;
      Pt[row][wn + lr]      = (__bf16)p0;
      Pt[row][wn + 16 + lr] = (__bf16)p1;
    }
    __syncthreads();
    #pragma unroll
    for (int half = 0; half < 2; ++half) {
      v16bf pa, kb;
      #pragma unroll
      for (int v = 0; v < 16; ++v) {
        int kk = (v < 8 ? v : v + 8) + hi8;
        pa[v] = Pt[wm + lr][half * 32 + kk];
        kb[v] = Kt[half * 32 + kk][dh0 + lr];
      }
      ctx = wmma_bf16(pa, kb, ctx);
    }
    __syncthreads();
  }

  #pragma unroll
  for (int r = 0; r < 8; ++r) {
    int row = q0 + wm + r + hi8;
    ctxb[(size_t)(b * L_ + row) * DC_ + h * DH_ + dh0 + lr] = (__bf16)ctx[r];
  }
}

// ---------------------------------------------------------------------------
// Coupling gate: y2 = sig(gpre)*tanh(x2) + (1-sig)*x2 -> bf16 right half of cat
// ---------------------------------------------------------------------------
__global__ __launch_bounds__(256) void couple_kernel(const float* __restrict__ gpre,
                                                     const float* __restrict__ src,
                                                     __bf16* __restrict__ catb) {
  size_t i = (size_t)blockIdx.x * blockDim.x + threadIdx.x;
  if (i >= (size_t)N_ * DC_) return;
  size_t n = i >> 9; int c = (int)(i & (DC_ - 1));
  float g = 1.f / (1.f + __expf(-gpre[i]));
  float x2 = src[n * D_ + DC_ + c];
  float y2 = g * tanhf(x2) + (1.f - g) * x2;
  catb[n * D_ + DC_ + c] = (__bf16)y2;
}

// ---------------------------------------------------------------------------
// LayerNorm of (X+Y) with affine; optional bf16 copy.
// ---------------------------------------------------------------------------
__global__ __launch_bounds__(256) void ln_kernel(const float* __restrict__ X,
                                                 const float* __restrict__ Y,
                                                 const float* __restrict__ gam,
                                                 const float* __restrict__ bet,
                                                 float* __restrict__ outF,
                                                 __bf16* __restrict__ outB) {
  __shared__ float red[256];
  const int row = blockIdx.x, t = threadIdx.x;
  const size_t base = (size_t)row * D_;
  float s = 0.f;
  for (int c = t; c < D_; c += 256) s += X[base + c] + Y[base + c];
  red[t] = s; __syncthreads();
  for (int k = 128; k > 0; k >>= 1) { if (t < k) red[t] += red[t + k]; __syncthreads(); }
  float mean = red[0] / D_;
  __syncthreads();
  float v = 0.f;
  for (int c = t; c < D_; c += 256) { float d = X[base + c] + Y[base + c] - mean; v += d * d; }
  red[t] = v; __syncthreads();
  for (int k = 128; k > 0; k >>= 1) { if (t < k) red[t] += red[t + k]; __syncthreads(); }
  float rstd = rsqrtf(red[0] / D_ + 1e-5f);
  for (int c = t; c < D_; c += 256) {
    float val = (X[base + c] + Y[base + c] - mean) * rstd * gam[c] + bet[c];
    outF[base + c] = val;
    if (outB) outB[base + c] = (__bf16)val;
  }
}

// ---------------------------------------------------------------------------
// Top-2 routing: softmax(h@Wg+bg), renormalized top-2 gates, mask, counts.
// ---------------------------------------------------------------------------
__global__ __launch_bounds__(64) void route_kernel(const float* __restrict__ hf,
                                                   const float* __restrict__ Wg,
                                                   const float* __restrict__ bg,
                                                   float* __restrict__ gi,
                                                   float* __restrict__ msk,
                                                   float* __restrict__ cnt) {
  __shared__ float part[E_][64];
  const int n = blockIdx.x, t = threadIdx.x;
  float acc[E_];
  #pragma unroll
  for (int e = 0; e < E_; ++e) acc[e] = 0.f;
  const float* xr = hf + (size_t)n * D_;
  for (int c = t; c < D_; c += 64) {
    float xv = xr[c];
    const float* wr = Wg + (size_t)c * E_;
    #pragma unroll
    for (int e = 0; e < E_; ++e) acc[e] += xv * wr[e];
  }
  #pragma unroll
  for (int e = 0; e < E_; ++e) part[e][t] = acc[e];
  __syncthreads();
  if (t == 0) {
    float gate[E_]; float mx = -1e30f;
    for (int e = 0; e < E_; ++e) {
      float s = bg[e];
      for (int q = 0; q < 64; ++q) s += part[e][q];
      gate[e] = s; mx = fmaxf(mx, s);
    }
    float den = 0.f;
    for (int e = 0; e < E_; ++e) { gate[e] = __expf(gate[e] - mx); den += gate[e]; }
    for (int e = 0; e < E_; ++e) gate[e] /= den;
    int i0 = 0;
    for (int e = 1; e < E_; ++e) if (gate[e] > gate[i0]) i0 = e;
    int i1 = (i0 == 0) ? 1 : 0;
    for (int e = 0; e < E_; ++e) if (e != i0 && gate[e] > gate[i1]) i1 = e;
    float s2 = gate[i0] + gate[i1];
    float w0 = gate[i0] / s2, w1 = gate[i1] / s2;
    for (int e = 0; e < E_; ++e) {
      gi[(size_t)e * N_ + n]  = (e == i0) ? w0 : ((e == i1) ? w1 : 0.f);
      msk[(size_t)e * N_ + n] = (e == i0 || e == i1) ? 1.f : 0.f;
    }
    atomicAdd(&cnt[i0], 1.f);
    atomicAdd(&cnt[i1], 1.f);
  }
}

// stats[j] = sum_n hh[n][j]*mask[n] / max(cnt,1)
__global__ __launch_bounds__(256) void stats_kernel(const float* __restrict__ hh,
                                                    const float* __restrict__ mk,
                                                    const float* __restrict__ cnt,
                                                    float* __restrict__ stats) {
  int j = blockIdx.x * blockDim.x + threadIdx.x;
  if (j >= DFF_) return;
  float s = 0.f;
  for (int n = 0; n < N_; ++n) s += hh[(size_t)n * DFF_ + j] * mk[n];
  stats[j] = s / fmaxf(cnt[0], 1.f);
}

// omega = softplus(stats@Wom + bom); phi = stats@Wph + bph
__global__ __launch_bounds__(256) void affine_kernel(const float* __restrict__ stats,
                                                     const float* __restrict__ Wom,
                                                     const float* __restrict__ bom,
                                                     const float* __restrict__ Wph,
                                                     const float* __restrict__ bph,
                                                     float* __restrict__ omega,
                                                     float* __restrict__ phi) {
  int j = blockIdx.x * blockDim.x + threadIdx.x;
  if (j >= DFF_) return;
  float so = bom[j], sp = bph[j];
  for (int k = 0; k < DFF_; ++k) {
    float sv = stats[k];
    so += sv * Wom[(size_t)k * DFF_ + j];
    sp += sv * Wph[(size_t)k * DFF_ + j];
  }
  omega[j] = (so > 20.f) ? so : log1pf(__expf(so));
  phi[j] = sp;
}

// t = (cnt>1) ? omega*hh+phi : hh ; a = exact-erf GELU -> bf16
__global__ __launch_bounds__(256) void act_kernel(const float* __restrict__ hh,
                                                  const float* __restrict__ omega,
                                                  const float* __restrict__ phi,
                                                  const float* __restrict__ cnt,
                                                  __bf16* __restrict__ ab) {
  size_t i = (size_t)blockIdx.x * blockDim.x + threadIdx.x;
  if (i >= (size_t)N_ * DFF_) return;
  int j = (int)(i & (DFF_ - 1));
  float t = hh[i];
  if (cnt[0] > 1.f) t = omega[j] * t + phi[j];
  float a = 0.5f * t * (1.f + erff(t * 0.70710678118654752f));
  ab[i] = (__bf16)a;
}

// ---------------------------------------------------------------------------
extern "C" void kernel_launch(void* const* d_in, const int* in_sizes, int n_in,
                              void* d_out, int out_size, void* d_ws, size_t ws_size,
                              hipStream_t stream) {
  (void)in_sizes; (void)n_in; (void)out_size; (void)ws_size;
  const float* src    = (const float*)d_in[0];
  const float* Wq     = (const float*)d_in[1];
  const float* bq     = (const float*)d_in[2];
  const float* Wk     = (const float*)d_in[3];
  const float* bk     = (const float*)d_in[4];
  const float* conv_w = (const float*)d_in[5];
  const float* conv_b = (const float*)d_in[6];
  const float* Wst    = (const float*)d_in[7];
  const float* bst    = (const float*)d_in[8];
  const float* Wout   = (const float*)d_in[9];
  const float* bout   = (const float*)d_in[10];
  const float* ln1_g  = (const float*)d_in[11];
  const float* ln1_b  = (const float*)d_in[12];
  const float* ln2_g  = (const float*)d_in[13];
  const float* ln2_b  = (const float*)d_in[14];
  const float* Wg     = (const float*)d_in[15];
  const float* bg     = (const float*)d_in[16];
  const float* W1     = (const float*)d_in[17];
  const float* b1     = (const float*)d_in[18];
  const float* Wom    = (const float*)d_in[19];
  const float* bom    = (const float*)d_in[20];
  const float* Wph    = (const float*)d_in[21];
  const float* bph    = (const float*)d_in[22];
  const float* W2     = (const float*)d_in[23];
  const float* b2     = (const float*)d_in[24];

  float* out2 = (float*)d_out;
  float* attn = out2 + (size_t)N_ * D_;

  char* wsb = (char*)d_ws;
  size_t off = 0;
  auto alloc = [&](size_t bytes) -> void* {
    void* p = wsb + off;
    off = (off + bytes + 255) & ~(size_t)255;
    return p;
  };
  __bf16* catb  = (__bf16*)alloc((size_t)N_ * D_ * 2);
  __bf16* Wqb   = (__bf16*)alloc((size_t)DC_ * DC_ * 2);
  __bf16* Wkb   = (__bf16*)alloc((size_t)DC_ * DC_ * 2);
  __bf16* Wstb  = (__bf16*)alloc((size_t)DC_ * DC_ * 2);
  __bf16* Woutb = (__bf16*)alloc((size_t)D_ * D_ * 2);
  __bf16* W1b   = (__bf16*)alloc((size_t)E_ * D_ * DFF_ * 2);
  __bf16* W2b   = (__bf16*)alloc((size_t)E_ * DFF_ * D_ * 2);
  __bf16* Qb    = (__bf16*)alloc((size_t)N_ * DC_ * 2);
  __bf16* Kb    = (__bf16*)alloc((size_t)N_ * DC_ * 2);
  float*  sal   = (float*)alloc((size_t)B_ * H_ * L_ * 4);
  __bf16* ctxb  = (__bf16*)alloc((size_t)N_ * DC_ * 2);
  float*  gpre  = (float*)alloc((size_t)N_ * DC_ * 4);
  float*  hpre  = (float*)alloc((size_t)N_ * D_ * 4);
  float*  hf    = (float*)alloc((size_t)N_ * D_ * 4);
  __bf16* hb    = (__bf16*)alloc((size_t)N_ * D_ * 2);
  float*  gi    = (float*)alloc((size_t)E_ * N_ * 4);
  float*  msk   = (float*)alloc((size_t)E_ * N_ * 4);
  float*  cnt   = (float*)alloc((size_t)E_ * 4);
  float*  stats = (float*)alloc((size_t)DFF_ * 4);
  float*  omega = (float*)alloc((size_t)DFF_ * 4);
  float*  phi   = (float*)alloc((size_t)DFF_ * 4);
  float*  hh    = (float*)alloc((size_t)N_ * DFF_ * 4);
  __bf16* ab    = (__bf16*)alloc((size_t)N_ * DFF_ * 2);
  float*  moe   = (float*)alloc((size_t)N_ * D_ * 4);

  auto cvt = [&](const float* x, __bf16* y, size_t n) {
    long blocks = (long)((n + 255) / 256);
    if (blocks > 65535) blocks = 65535;
    cvt_f32_bf16<<<(int)blocks, 256, 0, stream>>>(x, y, (long)n);
  };

  // ---- weight conversion (fresh each call: deterministic, graph-safe) ----
  cvt(Wq,   Wqb,   (size_t)DC_ * DC_);
  cvt(Wk,   Wkb,   (size_t)DC_ * DC_);
  cvt(Wst,  Wstb,  (size_t)DC_ * DC_);
  cvt(Wout, Woutb, (size_t)D_ * D_);
  cvt(W1,   W1b,   (size_t)E_ * D_ * DFF_);
  cvt(W2,   W2b,   (size_t)E_ * DFF_ * D_);

  pack_x1_kernel<<<(N_ * DC_ + 255) / 256, 256, 0, stream>>>(src, catb);
  sal_kernel<<<(B_ * H_ * L_ + 255) / 256, 256, 0, stream>>>(src, conv_w, conv_b, sal);

  // Q / K projections: [8192,512] x [512,512] -> bf16
  gemm_bf16_kernel<1><<<dim3(N_ / 128, DC_ / 64), 256, 0, stream>>>(
      catb, D_, Wqb, DC_, bq, nullptr, Qb, DC_, nullptr, N_, DC_, DC_);
  gemm_bf16_kernel<1><<<dim3(N_ / 128, DC_ / 64), 256, 0, stream>>>(
      catb, D_, Wkb, DC_, bk, nullptr, Kb, DC_, nullptr, N_, DC_, DC_);

  // fused attention: attn f32 to d_out, ctx bf16 to ws
  attn_kernel<<<dim3(L_ / 64, H_, B_), 256, 0, stream>>>(Qb, Kb, sal, attn, ctxb);

  // gate pre-activation: ctx @ Wst + bst
  gemm_bf16_kernel<0><<<dim3(N_ / 128, DC_ / 64), 256, 0, stream>>>(
      ctxb, DC_, Wstb, DC_, bst, gpre, nullptr, DC_, nullptr, N_, DC_, DC_);
  couple_kernel<<<(N_ * DC_ + 255) / 256, 256, 0, stream>>>(gpre, src, catb);

  // out = cat @ Wout + bout ; h = LN(src + out)
  gemm_bf16_kernel<0><<<dim3(N_ / 128, D_ / 64), 256, 0, stream>>>(
      catb, D_, Woutb, D_, bout, hpre, nullptr, D_, nullptr, N_, D_, D_);
  ln_kernel<<<N_, 256, 0, stream>>>(src, hpre, ln1_g, ln1_b, hf, hb);

  // ---- MoE ----
  hipMemsetAsync(cnt, 0, E_ * sizeof(float), stream);
  hipMemsetAsync(moe, 0, (size_t)N_ * D_ * sizeof(float), stream);
  route_kernel<<<N_, 64, 0, stream>>>(hf, Wg, bg, gi, msk, cnt);

  for (int e = 0; e < E_; ++e) {
    // hh = h @ W1[e] + b1[e]   [8192,4096] f32
    gemm_bf16_kernel<0><<<dim3(N_ / 128, DFF_ / 64), 256, 0, stream>>>(
        hb, D_, W1b + (size_t)e * D_ * DFF_, DFF_, b1 + (size_t)e * DFF_,
        hh, nullptr, DFF_, nullptr, N_, DFF_, D_);
    stats_kernel<<<DFF_ / 256, 256, 0, stream>>>(hh, msk + (size_t)e * N_, cnt + e, stats);
    affine_kernel<<<DFF_ / 256, 256, 0, stream>>>(
        stats, Wom + (size_t)e * DFF_ * DFF_, bom + (size_t)e * DFF_,
        Wph + (size_t)e * DFF_ * DFF_, bph + (size_t)e * DFF_, omega, phi);
    act_kernel<<<(int)(((size_t)N_ * DFF_ + 255) / 256), 256, 0, stream>>>(
        hh, omega, phi, cnt + e, ab);
    // moe += (a @ W2[e] + b2[e]) * gi[e]
    gemm_bf16_kernel<2><<<dim3(N_ / 128, D_ / 64), 256, 0, stream>>>(
        ab, DFF_, W2b + (size_t)e * DFF_ * D_, D_, b2 + (size_t)e * D_,
        moe, nullptr, D_, gi + (size_t)e * N_, N_, D_, DFF_);
  }

  // out2 = LN(h + moe)
  ln_kernel<<<N_, 256, 0, stream>>>(hf, moe, ln2_g, ln2_b, out2, nullptr);
}
